// MRM_26869315403793
// MI455X (gfx1250) — matrix operand order repaired and checked by reading, not compiled, not run on hardware
//
#include <hip/hip_runtime.h>

// ---------------------------------------------------------------------------
// CDNA5 (gfx1250) fused dual-branch channel-attention block.
// wave32, WMMA bf16 16x16x32, f32 accumulate.
//   K1: LayerNorm + 6 projection conv1x1 + both modulation levels (WMMA)
//   K2: depthwise 3x3 + modulation apply + q/k norms + per-head Gram (WMMA)
//   K3: softmax + fold po_w @ blockdiag(attn) into effective 64x64 weights
//   K4: conv1x1 with effective weights + residual (WMMA)
// ---------------------------------------------------------------------------

typedef __attribute__((ext_vector_type(16))) __bf16 v16bf;
typedef __attribute__((ext_vector_type(8)))  float  v8f;

constexpr long PIX  = 65536;   // 256*256 pixels per image
constexpr int  NB   = 4;       // batch
constexpr int  PCM  = 264;     // channel-major LDS pitch (bf16) - K2 staging
constexpr int  PPM  = 72;      // pixel-major LDS pitch (bf16)   - K1/K4 staging
constexpr int  LCM  = 64 * PCM;    // 64ch x 256px channel-major buffer
constexpr int  LPM  = 256 * PPM;   // 256px x 64ch pixel-major buffer
constexpr int  NTSTEP = 16 * PPM;  // B-fragment pointer step per N-tile

// ---------------- WMMA helpers --------------------------------------------

__device__ __forceinline__ v8f wmma_bf16(v16bf a, v16bf b, v8f c) {
  return __builtin_amdgcn_wmma_f32_16x16x32_bf16(
      /*neg_a=*/false, a, /*neg_b=*/false, b,
      /*c_mod=*/(short)0, c, /*reuse_a=*/false, /*reuse_b=*/false);
}

// Load 16 consecutive bf16 (2x ds_load_b128 / global_load_b128).
__device__ __forceinline__ v16bf load16(const __bf16* p) {
  v16bf v;
#pragma unroll
  for (int e = 0; e < 16; ++e) v[e] = p[e];
  return v;
}

// 16-bit A as loaded here (16x32): lane l holds row M=m0+(l&15); within-lane element
// e maps to K = k0 + hi16 + e, hi16 = (l>=16)*16.  One contiguous
// 16-element run -> vectorizes to 2x b128 loads.
__device__ __forceinline__ v16bf load_a_f32(const float* W, int m0, int k0, int lane) {
  v16bf a;
  const float* row = W + (m0 + (lane & 15)) * 64 + k0 + ((lane >> 4) ? 16 : 0);
#pragma unroll
  for (int e = 0; e < 8; ++e)  a[e] = (__bf16)row[e];
#pragma unroll
  for (int e = 8; e < 16; ++e) a[e] = (__bf16)row[e];
  return a;
}

__device__ __forceinline__ v16bf load_a_bf(const __bf16* W, int m0, int k0, int lane) {
  v16bf a;
  const __bf16* row = W + (m0 + (lane & 15)) * 64 + k0 + ((lane >> 4) ? 16 : 0);
#pragma unroll
  for (int e = 0; e < 8; ++e)  a[e] = row[e];
#pragma unroll
  for (int e = 8; e < 16; ++e) a[e] = row[e];
  return a;
}

// A fragment given a precomputed per-lane row pointer (one contiguous 16-elem run).
__device__ __forceinline__ v16bf load_a_runs(const __bf16* row) {
  v16bf a;
#pragma unroll
  for (int e = 0; e < 8; ++e)  a[e] = row[e];
#pragma unroll
  for (int e = 8; e < 16; ++e) a[e] = row[e];
  return a;
}

// C/D f32 layout: VGPR r, lane l -> M = r + (l>=16?8:0), N = l&15.
__device__ __forceinline__ void store_tile(__bf16* tensor, long planeBase, long pixBase,
                                           int m0, int n0, int lane, v8f c) {
  int n  = n0 + (lane & 15);
  int mb = m0 + ((lane >> 4) << 3);
#pragma unroll
  for (int r = 0; r < 8; ++r)
    tensor[(planeBase + mb + r) * PIX + pixBase + n] = (__bf16)c[r];
}

__device__ __forceinline__ void store_tile_bias(__bf16* tensor, long planeBase, long pixBase,
                                                int m0, int n0, int lane, v8f c,
                                                const float* bias) {
  int n  = n0 + (lane & 15);
  int mb = m0 + ((lane >> 4) << 3);
#pragma unroll
  for (int r = 0; r < 8; ++r)
    tensor[(planeBase + mb + r) * PIX + pixBase + n] = (__bf16)(c[r] + bias[mb + r]);
}

// Store to pixel-major LDS with bias + leaky relu (contiguous in r).
__device__ __forceinline__ void store_lds_act(__bf16* L, int m0, int n0, int lane, v8f c,
                                              const float* bias) {
  int n  = n0 + (lane & 15);
  int mb = m0 + ((lane >> 4) << 3);
  __bf16* dst = L + n * PPM + mb;
#pragma unroll
  for (int r = 0; r < 8; ++r) {
    float f = c[r] + bias[mb + r];
    f = (f > 0.f) ? f : 0.1f * f;   // leaky relu 0.1
    dst[r] = (__bf16)f;
  }
}

// ---------------- K1: LayerNorm + projections + modulation chains ----------
// One block per (image row, batch). 256 threads = 8 waves.
// LDS (pixel-major): XR YR XN YN MS1 MT1 MS2 MT2, each 256px x PPM bf16.
__global__ __launch_bounds__(256) void mrm_k1_ln_proj(
    const float* __restrict__ x, const float* __restrict__ y,
    const float* __restrict__ ln_w, const float* __restrict__ ln_b,
    const float* __restrict__ kv1_w, const float* __restrict__ q1_w,
    const float* __restrict__ kv2_w, const float* __restrict__ q2_w,
    const float* __restrict__ mm1_w, const float* __restrict__ mm1_b,
    const float* __restrict__ mm2_w, const float* __restrict__ mm2_b,
    __bf16* __restrict__ kv1pre, __bf16* __restrict__ q1pre,
    __bf16* __restrict__ kv2pre, __bf16* __restrict__ q2pre,
    __bf16* __restrict__ s1o, __bf16* __restrict__ t1o,
    __bf16* __restrict__ s2o, __bf16* __restrict__ t2o) {
  extern __shared__ __bf16 sm[];
  __bf16* XR  = sm;
  __bf16* YR  = XR + LPM;
  __bf16* XN  = YR + LPM;
  __bf16* YN  = XN + LPM;
  __bf16* MS1 = YN + LPM;
  __bf16* MT1 = MS1 + LPM;
  __bf16* MS2 = MT1 + LPM;
  __bf16* MT2 = MS2 + LPM;

  const int b   = blockIdx.y;
  const int row = blockIdx.x;
  const long pixBase = (long)row * 256;
  const int t = threadIdx.x;

  // ---- per-pixel LayerNorm over 64 channels (thread = pixel column) ----
  {
    const float* xp = x + (long)b * 64 * PIX + pixBase + t;
    const float* yp = y + (long)b * 64 * PIX + pixBase + t;
    __bf16* xr = XR + t * PPM;
    __bf16* yr = YR + t * PPM;
    float sx = 0.f, ssx = 0.f, sy = 0.f, ssy = 0.f;
    for (int c = 0; c < 64; ++c) {
      float vx = xp[(long)c * PIX];
      float vy = yp[(long)c * PIX];
      sx += vx; ssx += vx * vx;
      sy += vy; ssy += vy * vy;
      xr[c] = (__bf16)vx;
      yr[c] = (__bf16)vy;
    }
    float mux = sx * (1.f / 64.f), muy = sy * (1.f / 64.f);
    float ivx = rsqrtf(ssx * (1.f / 64.f) - mux * mux + 1e-5f);
    float ivy = rsqrtf(ssy * (1.f / 64.f) - muy * muy + 1e-5f);
    __bf16* xn = XN + t * PPM;
    __bf16* yn = YN + t * PPM;
    for (int c = 0; c < 64; ++c) {
      float w = ln_w[c], bb = ln_b[c];
      xn[c] = (__bf16)(((float)xr[c] - mux) * ivx * w + bb);
      yn[c] = (__bf16)(((float)yr[c] - muy) * ivy * w + bb);
    }
  }
  __syncthreads();

  const int lane = t & 31;
  const int wids = __builtin_amdgcn_readfirstlane(t >> 5);  // wave-uniform
  const int laneOff = (lane & 15) * PPM + ((lane >> 4) << 4); // B-frag lane base

  // ---- Round 1: 40 M-tiles, each wave owns 5; A loaded once per M-tile ----
  for (int mt = wids; mt < 40; mt += 8) {
    const float* Wp; const __bf16* src; int m0;
    if (mt < 8)       { Wp = kv1_w;          m0 = mt * 16;        src = XN; }
    else if (mt < 12) { Wp = q1_w;           m0 = (mt - 8) * 16;  src = YN; }
    else if (mt < 20) { Wp = kv2_w;          m0 = (mt - 12) * 16; src = YN; }
    else if (mt < 24) { Wp = q2_w;           m0 = (mt - 20) * 16; src = XN; }
    else if (mt < 28) { Wp = mm1_w;          m0 = (mt - 24) * 16; src = XR; }
    else if (mt < 32) { Wp = mm1_w + 2*4096; m0 = (mt - 28) * 16; src = XR; }
    else if (mt < 36) { Wp = mm2_w;          m0 = (mt - 32) * 16; src = YR; }
    else              { Wp = mm2_w + 2*4096; m0 = (mt - 36) * 16; src = YR; }

    const v16bf a0 = load_a_f32(Wp, m0, 0,  lane);
    const v16bf a1 = load_a_f32(Wp, m0, 32, lane);

    const __bf16* bp = src + laneOff;
    for (int nt = 0; nt < 16; ++nt, bp += NTSTEP) {
      v16bf b0 = load16(bp);        // issue all 4 ds_load_b128 first,
      v16bf b1 = load16(bp + 32);   // then back-to-back WMMAs
      v8f acc = {};
      acc = wmma_bf16(a0, b0, acc);
      acc = wmma_bf16(a1, b1, acc);
      if (mt < 8)       store_tile(kv1pre, (long)b * 128, pixBase, m0, nt * 16, lane, acc);
      else if (mt < 12) store_tile(q1pre,  (long)b * 64,  pixBase, m0, nt * 16, lane, acc);
      else if (mt < 20) store_tile(kv2pre, (long)b * 128, pixBase, m0, nt * 16, lane, acc);
      else if (mt < 24) store_tile(q2pre,  (long)b * 64,  pixBase, m0, nt * 16, lane, acc);
      else if (mt < 28) store_lds_act(MS1, m0, nt * 16, lane, acc, mm1_b);
      else if (mt < 32) store_lds_act(MT1, m0, nt * 16, lane, acc, mm1_b + 2 * 64);
      else if (mt < 36) store_lds_act(MS2, m0, nt * 16, lane, acc, mm2_b);
      else              store_lds_act(MT2, m0, nt * 16, lane, acc, mm2_b + 2 * 64);
    }
  }
  __syncthreads();

  // ---- Round 2: modulation level-2 -> s,t outputs (16 M-tiles, 2/wave) ----
  for (int mt = wids; mt < 16; mt += 8) {
    const float* Wp; const __bf16* src; __bf16* dst; const float* bias; int m0;
    if (mt < 4)       { Wp = mm1_w + 1*4096; src = MS1; dst = s1o; bias = mm1_b + 64;     m0 = mt * 16; }
    else if (mt < 8)  { Wp = mm1_w + 3*4096; src = MT1; dst = t1o; bias = mm1_b + 3 * 64; m0 = (mt - 4) * 16; }
    else if (mt < 12) { Wp = mm2_w + 1*4096; src = MS2; dst = s2o; bias = mm2_b + 64;     m0 = (mt - 8) * 16; }
    else              { Wp = mm2_w + 3*4096; src = MT2; dst = t2o; bias = mm2_b + 3 * 64; m0 = (mt - 12) * 16; }

    const v16bf a0 = load_a_f32(Wp, m0, 0,  lane);
    const v16bf a1 = load_a_f32(Wp, m0, 32, lane);
    const __bf16* bp = src + laneOff;
    for (int nt = 0; nt < 16; ++nt, bp += NTSTEP) {
      v16bf b0 = load16(bp);
      v16bf b1 = load16(bp + 32);
      v8f acc = {};
      acc = wmma_bf16(a0, b0, acc);
      acc = wmma_bf16(a1, b1, acc);
      store_tile_bias(dst, (long)b * 64, pixBase, m0, nt * 16, lane, acc, bias);
    }
  }
}

// ---------------- K2: dwconv3x3 + modulation apply + q/k reductions --------

__device__ __forceinline__ float dw9(const __bf16* plane, int row, int col,
                                     const float* w9) {
  float acc = 0.f;
#pragma unroll
  for (int dy = -1; dy <= 1; ++dy) {
    int r2 = row + dy;
    if ((unsigned)r2 >= 256u) continue;
#pragma unroll
    for (int dx = -1; dx <= 1; ++dx) {
      int c2 = col + dx;
      if ((unsigned)c2 >= 256u) continue;
      acc += w9[(dy + 1) * 3 + (dx + 1)] * (float)plane[r2 * 256 + c2];
    }
  }
  return acc;
}

__global__ __launch_bounds__(256) void mrm_k2_dw_reduce(
    const __bf16* __restrict__ kv1pre, const __bf16* __restrict__ q1pre,
    const __bf16* __restrict__ kv2pre, const __bf16* __restrict__ q2pre,
    const __bf16* __restrict__ s1, const __bf16* __restrict__ t1,
    const __bf16* __restrict__ s2, const __bf16* __restrict__ t2,
    const float* __restrict__ kvdw1, const float* __restrict__ qdw1,
    const float* __restrict__ kvdw2, const float* __restrict__ qdw2,
    __bf16* __restrict__ v1m, __bf16* __restrict__ v2m,
    float* __restrict__ qsq1, float* __restrict__ ksq1,
    float* __restrict__ qsq2, float* __restrict__ ksq2,
    float* __restrict__ G1, float* __restrict__ G2) {
  extern __shared__ __bf16 sm2[];
  __bf16* ldsQ = sm2;          // channel-major [64][PCM]
  __bf16* ldsK = ldsQ + LCM;

  const int b   = blockIdx.y;
  const int row = blockIdx.x;
  const int col = threadIdx.x;
  const long p  = (long)row * 256 + col;
  const int lane = col & 31;
  const int wids = __builtin_amdgcn_readfirstlane(col >> 5);

  for (int branch = 0; branch < 2; ++branch) {
    const __bf16* qpre  = (branch ? q2pre  : q1pre)  + (long)b * 64  * PIX;
    const __bf16* kvpre = (branch ? kv2pre : kv1pre) + (long)b * 128 * PIX;
    const __bf16* sP    = (branch ? s2 : s1) + (long)b * 64 * PIX;
    const __bf16* tP    = (branch ? t2 : t1) + (long)b * 64 * PIX;
    const float*  qdw   = branch ? qdw2  : qdw1;
    const float*  kvdw  = branch ? kvdw2 : kvdw1;
    __bf16*       vout  = (branch ? v2m : v1m) + (long)b * 64 * PIX;

    for (int c = 0; c < 64; ++c) {
      const __bf16* qplane = qpre  + (long)c * PIX;
      const __bf16* kplane = kvpre + (long)c * PIX;
      const __bf16* vplane = kvpre + (long)(c + 64) * PIX;
      if (c + 1 < 64)  // hint next channel plane into cache
        __builtin_prefetch(qpre + (long)(c + 1) * PIX + p, 0, 1);

      float qv = dw9(qplane, row, col, qdw + c * 9);
      float kv = dw9(kplane, row, col, kvdw + c * 9);
      float vv = dw9(vplane, row, col, kvdw + (c + 64) * 9);
      float sv = (float)sP[(long)c * PIX + p];
      float tv = (float)tP[(long)c * PIX + p];
      vout[(long)c * PIX + p] = (__bf16)(vv * (sv + 1.f) + tv);
      ldsQ[c * PCM + col] = (__bf16)qv;
      ldsK[c * PCM + col] = (__bf16)kv;
    }
    __syncthreads();

    // per-channel sum of squares (for q/k row norms): 128 entries
    float* qsq = (branch ? qsq2 : qsq1) + b * 64;
    float* ksq = (branch ? ksq2 : ksq1) + b * 64;
    if (col < 128) {
      int sel = col >> 6, c = col & 63;
      const __bf16* L = sel ? ldsK : ldsQ;
      float acc = 0.f;
      for (int pp = 0; pp < 256; ++pp) {
        float v = (float)L[c * PCM + pp];
        acc += v * v;
      }
      atomicAdd(sel ? &ksq[c] : &qsq[c], acc);
    }

    // per-head 8x8 Gram via WMMA: head-pair hp packs 2 heads into a 16x16
    // tile; K = 256 pixels split across 2 waves (4 WMMA each).
    {
      float* G = (branch ? G2 : G1) + b * 512;
      int hp   = wids & 3;        // head pair 0..3
      int half = wids >> 2;       // pixel half 0..1
      // per-lane base pointers (A: M=q channel / K=pixel, B: N=k channel)
      const __bf16* rowA = ldsQ + (hp * 16 + (lane & 15)) * PCM
                               + half * 128 + ((lane >> 4) ? 16 : 0);
      const __bf16* colB = ldsK + (hp * 16 + (lane & 15)) * PCM
                               + half * 128 + ((lane >> 4) << 4);
      v8f acc = {};
#pragma unroll
      for (int kt = 0; kt < 4; ++kt) {
        v16bf a  = load_a_runs(rowA + kt * 32);
        v16bf bb = load16(colB + kt * 32);
        acc = wmma_bf16(a, bb, acc);
      }
      int hi = lane >> 4;        // which head of the pair this lane's M is in
      int nk = lane & 15;        // k channel within pair
      if ((nk >> 3) == hi) {     // keep same-head 8x8 blocks only
        int h = hp * 2 + hi, d = nk & 7;
#pragma unroll
        for (int r = 0; r < 8; ++r)
          atomicAdd(&G[h * 64 + r * 8 + d], acc[r]);
      }
    }
    __syncthreads();  // before branch 2 reuses LDS
  }
}

// ---------------- K3: finalize attention, fold into effective weights ------

__global__ __launch_bounds__(256) void mrm_k3_attn(
    const float* __restrict__ temp,
    const float* __restrict__ po1, const float* __restrict__ po2,
    const float* __restrict__ qsq1, const float* __restrict__ ksq1,
    const float* __restrict__ qsq2, const float* __restrict__ ksq2,
    const float* __restrict__ G1, const float* __restrict__ G2,
    __bf16* __restrict__ Weff) {
  __shared__ float attnS[512];
  __shared__ float qn[64], kn[64];
  const int t = threadIdx.x;

  for (int combo = 0; combo < 8; ++combo) {
    int branch = combo >> 2, b = combo & 3;
    const float* G   = (branch ? G2 : G1) + b * 512;
    const float* qsq = (branch ? qsq2 : qsq1) + b * 64;
    const float* ksq = (branch ? ksq2 : ksq1) + b * 64;
    if (t < 64) {
      qn[t] = fmaxf(sqrtf(qsq[t]), 1e-12f);
      kn[t] = fmaxf(sqrtf(ksq[t]), 1e-12f);
    }
    __syncthreads();
#pragma unroll
    for (int i = 0; i < 2; ++i) {
      int idx = t * 2 + i;
      int h = idx >> 6, c = (idx >> 3) & 7, d = idx & 7;
      attnS[idx] = G[idx] / (qn[h * 8 + c] * kn[h * 8 + d]) * temp[h];
    }
    __syncthreads();
    if (t < 64) {  // softmax over d for row (h,c)
      float* rowp = &attnS[(t >> 3) * 64 + (t & 7) * 8];
      float mx = rowp[0];
      for (int d = 1; d < 8; ++d) mx = fmaxf(mx, rowp[d]);
      float sum = 0.f;
      for (int d = 0; d < 8; ++d) { rowp[d] = __expf(rowp[d] - mx); sum += rowp[d]; }
      float inv = 1.f / sum;
      for (int d = 0; d < 8; ++d) rowp[d] *= inv;
    }
    __syncthreads();
    // Weff[o][h*8+d] = sum_c po[o][h*8+c] * attn[h][c][d]
    const float* po = branch ? po2 : po1;
    __bf16* Wf = Weff + combo * 4096;
#pragma unroll
    for (int r = 0; r < 16; ++r) {
      int idx = t * 16 + r;
      int o = idx >> 6, j = idx & 63, h = j >> 3, d = j & 7;
      float acc = 0.f;
      for (int c = 0; c < 8; ++c)
        acc += po[o * 64 + h * 8 + c] * attnS[h * 64 + c * 8 + d];
      Wf[idx] = (__bf16)acc;
    }
    __syncthreads();
  }
}

// ---------------- K4: output conv1x1 with effective weights + residual -----

__global__ __launch_bounds__(256) void mrm_k4_out(
    const __bf16* __restrict__ v1m, const __bf16* __restrict__ v2m,
    const __bf16* __restrict__ Weff,
    const float* __restrict__ x, const float* __restrict__ y,
    float* __restrict__ out) {
  extern __shared__ __bf16 ldsV[];  // pixel-major [256][PPM]
  const int branch = blockIdx.z;
  const int b      = blockIdx.y;
  const int row    = blockIdx.x;
  const long pixBase = (long)row * 256;
  const int t = threadIdx.x;

  const __bf16* v = (branch ? v2m : v1m) + (long)b * 64 * PIX;
  {
    __bf16* dstL = ldsV + t * PPM;
    for (int c = 0; c < 64; ++c)
      dstL[c] = v[(long)c * PIX + pixBase + t];
  }
  __syncthreads();

  const __bf16* Wf  = Weff + (branch * 4 + b) * 4096;
  const float*  res = (branch ? x : y) + (long)b * 64 * PIX;  // out1+=y, out2+=x
  float*        dst = out + ((long)branch * NB + b) * 64 * PIX;

  const int lane = t & 31;
  const int wids = __builtin_amdgcn_readfirstlane(t >> 5);
  const int mt = wids & 3;             // M-tile (4 of them)
  const int ntb = (wids >> 2) * 8;     // N-tile range split across wave halves
  const int laneOff = (lane & 15) * PPM + ((lane >> 4) << 4);

  const v16bf a0 = load_a_bf(Wf, mt * 16, 0,  lane);
  const v16bf a1 = load_a_bf(Wf, mt * 16, 32, lane);

  const __bf16* bp = ldsV + laneOff + ntb * NTSTEP;
  for (int nt = ntb; nt < ntb + 8; ++nt, bp += NTSTEP) {
    v16bf b0 = load16(bp);
    v16bf b1 = load16(bp + 32);
    v8f acc = {};
    acc = wmma_bf16(a0, b0, acc);
    acc = wmma_bf16(a1, b1, acc);
    int n  = nt * 16 + (lane & 15);
    int mb = mt * 16 + ((lane >> 4) << 3);
#pragma unroll
    for (int r = 0; r < 8; ++r) {
      long off = (long)(mb + r) * PIX + pixBase + n;
      dst[off] = acc[r] + res[off];
    }
  }
}

// ---------------- launch ----------------------------------------------------

extern "C" void kernel_launch(void* const* d_in, const int* in_sizes, int n_in,
                              void* d_out, int out_size, void* d_ws, size_t ws_size,
                              hipStream_t stream) {
  (void)in_sizes; (void)n_in; (void)out_size; (void)ws_size;
  const float* x     = (const float*)d_in[0];
  const float* y     = (const float*)d_in[1];
  const float* ln_w  = (const float*)d_in[2];
  const float* ln_b  = (const float*)d_in[3];
  const float* temp  = (const float*)d_in[4];
  const float* kv1_w = (const float*)d_in[5];
  const float* kvdw1 = (const float*)d_in[6];
  const float* q1_w  = (const float*)d_in[7];
  const float* qdw1  = (const float*)d_in[8];
  const float* po1_w = (const float*)d_in[9];
  const float* kv2_w = (const float*)d_in[10];
  const float* kvdw2 = (const float*)d_in[11];
  const float* q2_w  = (const float*)d_in[12];
  const float* qdw2  = (const float*)d_in[13];
  const float* po2_w = (const float*)d_in[14];
  const float* mm1_w = (const float*)d_in[15];
  const float* mm1_b = (const float*)d_in[16];
  const float* mm2_w = (const float*)d_in[17];
  const float* mm2_b = (const float*)d_in[18];

  // ---- workspace layout (bytes) ----
  constexpr size_t SZ64  = (size_t)NB * 64  * PIX * 2;  // 33,554,432
  constexpr size_t SZ128 = (size_t)NB * 128 * PIX * 2;  // 67,108,864
  char* w8 = (char*)d_ws;
  size_t off = 0;
  __bf16* kv1pre = (__bf16*)(w8 + off); off += SZ128;
  __bf16* q1pre  = (__bf16*)(w8 + off); off += SZ64;
  __bf16* kv2pre = (__bf16*)(w8 + off); off += SZ128;
  __bf16* q2pre  = (__bf16*)(w8 + off); off += SZ64;
  __bf16* s1o    = (__bf16*)(w8 + off); off += SZ64;
  __bf16* t1o    = (__bf16*)(w8 + off); off += SZ64;
  __bf16* s2o    = (__bf16*)(w8 + off); off += SZ64;
  __bf16* t2o    = (__bf16*)(w8 + off); off += SZ64;
  __bf16* v1m    = (__bf16*)(w8 + off); off += SZ64;
  __bf16* v2m    = (__bf16*)(w8 + off); off += SZ64;
  size_t redOff = off;
  float* qsq1 = (float*)(w8 + off); off += NB * 64 * 4;
  float* ksq1 = (float*)(w8 + off); off += NB * 64 * 4;
  float* qsq2 = (float*)(w8 + off); off += NB * 64 * 4;
  float* ksq2 = (float*)(w8 + off); off += NB * 64 * 4;
  float* G1   = (float*)(w8 + off); off += NB * 512 * 4;
  float* G2   = (float*)(w8 + off); off += NB * 512 * 4;
  size_t redBytes = off - redOff;
  __bf16* Weff = (__bf16*)(w8 + off); off += 8 * 4096 * 2;

  // zero the atomic accumulators each call (graph-capture safe)
  hipMemsetAsync(w8 + redOff, 0, redBytes, stream);

  const size_t smem1 = 8 * (size_t)LPM * sizeof(__bf16);  // 294,912 B (<320 KB LDS)
  const size_t smem2 = 2 * (size_t)LCM * sizeof(__bf16);  //  67,584 B
  const size_t smem4 = 1 * (size_t)LPM * sizeof(__bf16);  //  36,864 B

  dim3 g1(256, NB);
  mrm_k1_ln_proj<<<g1, 256, smem1, stream>>>(
      x, y, ln_w, ln_b, kv1_w, q1_w, kv2_w, q2_w,
      mm1_w, mm1_b, mm2_w, mm2_b,
      kv1pre, q1pre, kv2pre, q2pre, s1o, t1o, s2o, t2o);

  dim3 g2(256, NB);
  mrm_k2_dw_reduce<<<g2, 256, smem2, stream>>>(
      kv1pre, q1pre, kv2pre, q2pre, s1o, t1o, s2o, t2o,
      kvdw1, qdw1, kvdw2, qdw2, v1m, v2m,
      qsq1, ksq1, qsq2, ksq2, G1, G2);

  mrm_k3_attn<<<1, 256, 0, stream>>>(
      temp, po1_w, po2_w, qsq1, ksq1, qsq2, ksq2, G1, G2, Weff);

  dim3 g4(256, NB, 2);
  mrm_k4_out<<<g4, 256, smem4, stream>>>(
      v1m, v2m, Weff, x, y, (float*)d_out);
}
